// Block_25271587569733
// MI455X (gfx1250) — compile-verified, matrix operand order checked by reading, not run on hardware
//
#include <hip/hip_runtime.h>
#include <hip/hip_bf16.h>

// ---------------------------------------------------------------------------
// Transformer block forward for MI455X (gfx1250), wave32 + WMMA bf16,
// async global->LDS staging (ASYNCcnt) for GEMM tiles.
//   B=4, Nx=1024, Nc=512, DIM=1024, HEADS=16, HEAD_DIM=64, HIDDEN=4096
// ---------------------------------------------------------------------------

typedef __bf16 bf16;
typedef __attribute__((ext_vector_type(16))) __bf16 v16bf;
typedef __attribute__((ext_vector_type(8)))  float  v8f;

#define NB   4
#define NX   1024
#define NC   512
#define CDIM 1024
#define NH   16
#define HD   64
#define HID  4096

// ---------------------------------------------------------------------------
// WMMA helper: D = A(16x32 bf16) * B(32x16 bf16) + C(16x16 f32)
// ---------------------------------------------------------------------------
__device__ __forceinline__ v8f wmma_bf16(v16bf a, v16bf b, v8f c) {
    return __builtin_amdgcn_wmma_f32_16x16x32_bf16(
        /*neg_a=*/false, a, /*neg_b=*/false, b,
        /*c_mod=*/(short)0, c, /*reuse_a=*/false, /*reuse_b=*/false);
}

// CDNA5 async DMA: global -> LDS, 16 bytes per lane, tracked by ASYNCcnt.
// VDST = per-lane LDS byte address (low 32 bits of the generic pointer:
// flat-LDS aperture keeps the DS offset in addr[31:0]).
__device__ __forceinline__ void async_load_b128(void* lds, const void* g) {
    unsigned           l = (unsigned)(unsigned long long)lds;
    unsigned long long a = (unsigned long long)g;
    asm volatile("global_load_async_to_lds_b128 %0, %1, off"
                 :: "v"(l), "v"(a) : "memory");
}
__device__ __forceinline__ void wait_async0() {
    asm volatile("s_wait_asynccnt 0x0" ::: "memory");
}

// Load one 16x32 A/B fragment.  Per ISA 7.12.2 (16-bit A 16x32):
//   lane 0-15 : row = lane,    K in {0..7, 16..23}
//   lane 16-31: row = lane-16, K in {8..15, 24..31}
// base points at (row0, k0) of the fragment; stride is row stride in elems.
__device__ __forceinline__ v16bf load_frag(const bf16* base, int stride,
                                           int lane, int kOff) {
    const int r  = lane & 15;
    const int kb = (lane >> 4) << 3;          // 0 or 8
    const bf16* p = base + (size_t)r * stride + kOff + kb;
    union { v16bf v; uint4 u[2]; } f;
    f.u[0] = *(const uint4*)(p);
    f.u[1] = *(const uint4*)(p + 16);
    return f.v;
}

// ---------------------------------------------------------------------------
// GEMM: C[M,N] = act(A[M,K](bf16) * W[N,K](f32)^T + bias) (+ residual)
// Block tile 128x128, 256 threads (8 waves), wave tile 64(M) x 32(N).
// A tile staged with global_load_async_to_lds_b128; W converted f32->bf16
// through VGPRs while the async DMA runs; next slab prefetched to L2.
// ---------------------------------------------------------------------------
template<bool BIAS, bool GELU, bool RES, typename OutT>
__global__ __launch_bounds__(256)
void gemm_xwT(const bf16* __restrict__ A, int lda,
              const float* __restrict__ W,           // [N,K] row-major
              OutT* __restrict__ C, int ldc,
              const float* __restrict__ bias,
              const float* __restrict__ res, int ldr,
              int K) {
    __shared__ __align__(16) bf16 As[128][72];
    __shared__ __align__(16) bf16 Bs[128][72];

    const int tid  = threadIdx.x;
    const int lane = tid & 31;
    const int wid  = tid >> 5;
    const int wm   = (wid >> 2) * 64;   // wave row offset within tile (0,64)
    const int wn   = (wid & 3) * 32;    // wave col offset (0,32,64,96)
    const int bM   = blockIdx.y * 128;
    const int bN   = blockIdx.x * 128;

    v8f acc[4][2];
    v8f vz = {};
    #pragma unroll
    for (int i = 0; i < 4; i++)
        #pragma unroll
        for (int j = 0; j < 2; j++) acc[i][j] = vz;

    for (int k0 = 0; k0 < K; k0 += 64) {
        // ---- stage A tile (128 x 64 bf16) via async DMA to LDS ----
        #pragma unroll
        for (int i = 0; i < 4; i++) {
            int c   = tid + i * 256;            // 0..1023
            int r   = c >> 3;
            int col = (c & 7) * 8;
            async_load_b128(&As[r][col],
                            A + (size_t)(bM + r) * lda + k0 + col);
        }
        // ---- stage W tile (128 x 64 f32 -> bf16) through VGPRs ----
        #pragma unroll
        for (int i = 0; i < 8; i++) {
            int c   = tid + i * 256;            // 0..2047
            int r   = c >> 4;
            int col = (c & 15) * 4;
            float4 w4 = *(const float4*)(W + (size_t)(bN + r) * K + k0 + col);
            bf16* d = &Bs[r][col];
            d[0] = (bf16)w4.x; d[1] = (bf16)w4.y;
            d[2] = (bf16)w4.z; d[3] = (bf16)w4.w;
        }
        // ---- prefetch next K slab into cache while we compute ----
        if (k0 + 64 < K) {
            int pr = tid & 127;
            __builtin_prefetch(W + (size_t)(bN + pr) * K + k0 + 64 +
                               ((tid >> 7) << 5), 0, 0);
            if (tid < 128)
                __builtin_prefetch(A + (size_t)(bM + pr) * lda + k0 + 64, 0, 0);
        }
        wait_async0();
        __syncthreads();

        #pragma unroll
        for (int ks = 0; ks < 64; ks += 32) {
            v16bf af[4], bfg[2];
            #pragma unroll
            for (int mi = 0; mi < 4; mi++)
                af[mi] = load_frag(&As[wm + mi * 16][0], 72, lane, ks);
            #pragma unroll
            for (int ni = 0; ni < 2; ni++)
                bfg[ni] = load_frag(&Bs[wn + ni * 16][0], 72, lane, ks);
            #pragma unroll
            for (int mi = 0; mi < 4; mi++)
                #pragma unroll
                for (int ni = 0; ni < 2; ni++)
                    acc[mi][ni] = wmma_bf16(af[mi], bfg[ni], acc[mi][ni]);
        }
        __syncthreads();
    }

    // ---- epilogue: C layout m = r + 8*(lane>=16), n = lane&15 ----
    const int halfm = (lane >> 4) << 3;
    const int nl    = lane & 15;
    #pragma unroll
    for (int mi = 0; mi < 4; mi++)
        #pragma unroll
        for (int ni = 0; ni < 2; ni++)
            #pragma unroll
            for (int r = 0; r < 8; r++) {
                int gm = bM + wm + mi * 16 + r + halfm;
                int gn = bN + wn + ni * 16 + nl;
                float v = acc[mi][ni][r];
                if (BIAS) v += bias[gn];
                if (GELU) v = 0.5f * v * (1.0f + erff(v * 0.70710678118654752f));
                if (RES)  v += res[(size_t)gm * ldr + gn];
                C[(size_t)gm * ldc + gn] = (OutT)v;
            }
}

// ---------------------------------------------------------------------------
// Fused flash attention, one (b, head, 64-query tile) per block, 4 waves.
// Each wave owns 16 query rows. Vt is pre-transposed: [B][H][64][Nk] bf16.
// ---------------------------------------------------------------------------
template<bool HAS_MASK>
__global__ __launch_bounds__(128)
void flash_attn(const bf16* __restrict__ Q,  int ldq,
                const bf16* __restrict__ Km, int ldk,
                const bf16* __restrict__ Vt,
                const int*  __restrict__ mask,
                bf16* __restrict__ Out, int ldo,
                int Nq, int Nk, float scale) {
    __shared__ __align__(16) bf16 Pls[4][16][72];  // wave-private P tiles

    const int tid  = threadIdx.x;
    const int lane = tid & 31;
    const int w    = tid >> 5;
    const int h    = blockIdx.y;
    const int b    = blockIdx.z;
    const int q0   = blockIdx.x * 64 + w * 16;
    const int nl   = lane & 15;
    const int halfm = (lane >> 4) << 3;

    const bf16* Qb = Q  + ((size_t)b * Nq) * ldq + h * HD;
    const bf16* Kb = Km + ((size_t)b * Nk) * ldk + h * HD;
    const bf16* Vb = Vt + ((size_t)(b * NH + h) * HD) * Nk;
    bf16*       Ob = Out + ((size_t)b * Nq) * ldo + h * HD;
    const int*  mb = HAS_MASK ? (mask + (size_t)b * Nk) : nullptr;

    // Q fragments (held for the whole key loop)
    v16bf qf0 = load_frag(Qb + (size_t)q0 * ldq, ldq, lane, 0);
    v16bf qf1 = load_frag(Qb + (size_t)q0 * ldq, ldq, lane, 32);

    v8f o[4];
    v8f vz = {};
    #pragma unroll
    for (int f = 0; f < 4; f++) o[f] = vz;
    float rm[8], rs[8];
    #pragma unroll
    for (int r = 0; r < 8; r++) { rm[r] = -1e30f; rs[r] = 0.0f; }

    for (int kblk = 0; kblk < Nk; kblk += 64) {
        // ---- S = Q * K^T (16 x 64 per wave) ----
        v8f s[4];
        #pragma unroll
        for (int nb = 0; nb < 4; nb++) {
            v8f c = vz;
            v16bf bk0 = load_frag(Kb + (size_t)(kblk + nb * 16) * ldk, ldk, lane, 0);
            c = wmma_bf16(qf0, bk0, c);
            v16bf bk1 = load_frag(Kb + (size_t)(kblk + nb * 16) * ldk, ldk, lane, 32);
            c = wmma_bf16(qf1, bk1, c);
            s[nb] = c;
        }
        // ---- scale + codebook mask bias ----
        #pragma unroll
        for (int nb = 0; nb < 4; nb++) {
            float mbias = 0.0f;
            if (HAS_MASK)
                mbias = (mb[kblk + nb * 16 + nl] != 0) ? 0.0f : -10000.0f;
            #pragma unroll
            for (int r = 0; r < 8; r++)
                s[nb][r] = s[nb][r] * scale + mbias;
        }
        // ---- online softmax: new row max (reduce across 16-lane half) ----
        float nm[8];
        #pragma unroll
        for (int r = 0; r < 8; r++) {
            float m = rm[r];
            #pragma unroll
            for (int nb = 0; nb < 4; nb++) m = fmaxf(m, s[nb][r]);
            m = fmaxf(m, __shfl_xor(m, 1));
            m = fmaxf(m, __shfl_xor(m, 2));
            m = fmaxf(m, __shfl_xor(m, 4));
            m = fmaxf(m, __shfl_xor(m, 8));
            nm[r] = m;
        }
        #pragma unroll
        for (int r = 0; r < 8; r++) {
            float alpha = __expf(rm[r] - nm[r]);
            rm[r] = nm[r];
            rs[r] *= alpha;
            #pragma unroll
            for (int f = 0; f < 4; f++) o[f][r] *= alpha;
        }
        // ---- P = exp(S - max), accumulate row sums ----
        #pragma unroll
        for (int r = 0; r < 8; r++) {
            float srow = 0.0f;
            #pragma unroll
            for (int nb = 0; nb < 4; nb++) {
                float p = __expf(s[nb][r] - rm[r]);
                s[nb][r] = p;
                srow += p;
            }
            srow += __shfl_xor(srow, 1);
            srow += __shfl_xor(srow, 2);
            srow += __shfl_xor(srow, 4);
            srow += __shfl_xor(srow, 8);
            rs[r] += srow;
        }
        // ---- C-layout -> A-layout via wave-private LDS ----
        #pragma unroll
        for (int nb = 0; nb < 4; nb++)
            #pragma unroll
            for (int r = 0; r < 8; r++)
                Pls[w][r + halfm][nb * 16 + nl] = (bf16)s[nb][r];
        v16bf pf0 = load_frag(&Pls[w][0][0], 72, lane, 0);
        v16bf pf1 = load_frag(&Pls[w][0][0], 72, lane, 32);
        // ---- O += P * V ----
        #pragma unroll
        for (int f = 0; f < 4; f++) {
            v16bf bv0 = load_frag(Vb + (size_t)(f * 16) * Nk + kblk, Nk, lane, 0);
            o[f] = wmma_bf16(pf0, bv0, o[f]);
            v16bf bv1 = load_frag(Vb + (size_t)(f * 16) * Nk + kblk, Nk, lane, 32);
            o[f] = wmma_bf16(pf1, bv1, o[f]);
        }
    }
    // ---- normalize + store bf16 ----
    #pragma unroll
    for (int r = 0; r < 8; r++) {
        float inv = 1.0f / rs[r];
        int gm = q0 + r + halfm;
        #pragma unroll
        for (int f = 0; f < 4; f++)
            Ob[(size_t)gm * ldo + f * 16 + nl] = (bf16)(o[f][r] * inv);
    }
}

// ---------------------------------------------------------------------------
// LayerNorm over 1024-elem rows -> bf16. One block (256 thr) per row.
// ---------------------------------------------------------------------------
__global__ __launch_bounds__(256)
void layernorm_bf16(const float* __restrict__ x, const float* __restrict__ w,
                    const float* __restrict__ bb, bf16* __restrict__ out) {
    __shared__ float s1[256], s2[256];
    const int row = blockIdx.x, tid = threadIdx.x;
    const float* xr = x + (size_t)row * CDIM;
    float4 v = ((const float4*)xr)[tid];
    s1[tid] = v.x + v.y + v.z + v.w;
    s2[tid] = v.x * v.x + v.y * v.y + v.z * v.z + v.w * v.w;
    __syncthreads();
    for (int off = 128; off > 0; off >>= 1) {
        if (tid < off) { s1[tid] += s1[tid + off]; s2[tid] += s2[tid + off]; }
        __syncthreads();
    }
    float mu   = s1[0] * (1.0f / CDIM);
    float var  = s2[0] * (1.0f / CDIM) - mu * mu;
    float rstd = rsqrtf(var + 1e-5f);
    bf16* orow = out + (size_t)row * CDIM;
    int i0 = tid * 4;
    float xs[4] = { v.x, v.y, v.z, v.w };
    #pragma unroll
    for (int j = 0; j < 4; j++)
        orow[i0 + j] = (bf16)((xs[j] - mu) * rstd * w[i0 + j] + bb[i0 + j]);
}

// ---------------------------------------------------------------------------
__global__ void cast_f32_bf16(const float* __restrict__ in,
                              bf16* __restrict__ out, int n) {
    int i = blockIdx.x * 256 + threadIdx.x;
    if (i < n) out[i] = (bf16)in[i];
}

// v: logical [B][Nk][rowStride] slice (bf16) -> vt: [B][H][64][Nk]
__global__ void transpose_v_kernel(const bf16* __restrict__ v, int rs,
                                   bf16* __restrict__ vt, int Nk) {
    int i = blockIdx.x * 256 + threadIdx.x;
    int total = NB * NH * HD * Nk;
    if (i >= total) return;
    int n = i % Nk;
    int d = (i / Nk) & (HD - 1);
    int h = (i / (Nk * HD)) & (NH - 1);
    int b = i / (Nk * HD * NH);
    vt[i] = v[((size_t)(b * Nk + n)) * rs + h * HD + d];
}

// ---------------------------------------------------------------------------
extern "C" void kernel_launch(void* const* d_in, const int* in_sizes, int n_in,
                              void* d_out, int out_size, void* d_ws, size_t ws_size,
                              hipStream_t stream) {
    const float* x     = (const float*)d_in[0];
    const float* cbemb = (const float*)d_in[1];
    const int*   cmask = (const int*)  d_in[2];
    const float* ln0w  = (const float*)d_in[3];
    const float* ln0b  = (const float*)d_in[4];
    const float* qkv_w = (const float*)d_in[5];
    const float* sa_w  = (const float*)d_in[6];
    const float* ln1w  = (const float*)d_in[7];
    const float* ln1b  = (const float*)d_in[8];
    const float* q_w   = (const float*)d_in[9];
    const float* k_w   = (const float*)d_in[10];
    const float* v_w   = (const float*)d_in[11];
    const float* ca_w  = (const float*)d_in[12];
    const float* ln2w  = (const float*)d_in[13];
    const float* ln2b  = (const float*)d_in[14];
    const float* fc1_w = (const float*)d_in[15];
    const float* fc1_b = (const float*)d_in[16];
    const float* fc2_w = (const float*)d_in[17];
    const float* fc2_b = (const float*)d_in[18];
    float* out = (float*)d_out;

    char* ws = (char*)d_ws;
    bf16*  XN  = (bf16*) (ws);                       //  8 MB  LN outputs
    bf16*  QKV = (bf16*) (ws + (size_t)( 8u << 20)); // 24 MB  qkv (bf16)
    bf16*  VT1 = (bf16*) (ws + (size_t)(32u << 20)); //  8 MB  V^T self
    bf16*  ATT = (bf16*) (ws + (size_t)(40u << 20)); //  8 MB  attn outs
    float* X1  = (float*)(ws + (size_t)(48u << 20)); // 16 MB  residual stream
    bf16*  CB  = (bf16*) (ws + (size_t)(64u << 20)); //  4 MB  codebook bf16
    bf16*  QC  = (bf16*) (ws + (size_t)(68u << 20)); //  8 MB  cross q
    bf16*  KC  = (bf16*) (ws + (size_t)(76u << 20)); //  4 MB  cross k
    bf16*  VC  = (bf16*) (ws + (size_t)(80u << 20)); //  4 MB  cross v
    bf16*  VT2 = (bf16*) (ws + (size_t)(84u << 20)); //  4 MB  V^T cross
    bf16*  HB  = QKV;   // 32 MB MLP hidden reuses QKV+VT1 (dead by then)

    const int Mx = NB * NX;   // 4096
    const int Mc = NB * NC;   // 2048
    dim3 thr256(256), thr128(128);

    // --- self attention branch ---
    layernorm_bf16<<<Mx, thr256, 0, stream>>>(x, ln0w, ln0b, XN);
    gemm_xwT<false,false,false,bf16><<<dim3(3 * CDIM / 128, Mx / 128), thr256, 0, stream>>>(
        XN, CDIM, qkv_w, QKV, 3 * CDIM, nullptr, nullptr, 0, CDIM);
    transpose_v_kernel<<<(NB * NH * HD * NX) / 256, thr256, 0, stream>>>(
        QKV + 2 * CDIM, 3 * CDIM, VT1, NX);
    flash_attn<false><<<dim3(NX / 64, NH, NB), thr128, 0, stream>>>(
        QKV, 3 * CDIM, QKV + CDIM, 3 * CDIM, VT1, nullptr, ATT, CDIM,
        NX, NX, 0.125f);
    gemm_xwT<false,false,true,float><<<dim3(CDIM / 128, Mx / 128), thr256, 0, stream>>>(
        ATT, CDIM, sa_w, X1, CDIM, nullptr, x, CDIM, CDIM);

    // --- cross attention branch ---
    layernorm_bf16<<<Mx, thr256, 0, stream>>>(X1, ln1w, ln1b, XN);
    gemm_xwT<false,false,false,bf16><<<dim3(CDIM / 128, Mx / 128), thr256, 0, stream>>>(
        XN, CDIM, q_w, QC, CDIM, nullptr, nullptr, 0, CDIM);
    cast_f32_bf16<<<(NB * NC * CDIM) / 256, thr256, 0, stream>>>(
        cbemb, CB, NB * NC * CDIM);
    gemm_xwT<false,false,false,bf16><<<dim3(CDIM / 128, Mc / 128), thr256, 0, stream>>>(
        CB, CDIM, k_w, KC, CDIM, nullptr, nullptr, 0, CDIM);
    gemm_xwT<false,false,false,bf16><<<dim3(CDIM / 128, Mc / 128), thr256, 0, stream>>>(
        CB, CDIM, v_w, VC, CDIM, nullptr, nullptr, 0, CDIM);
    transpose_v_kernel<<<(NB * NH * HD * NC) / 256, thr256, 0, stream>>>(
        VC, CDIM, VT2, NC);
    flash_attn<true><<<dim3(NX / 64, NH, NB), thr128, 0, stream>>>(
        QC, CDIM, KC, CDIM, VT2, cmask, ATT, CDIM, NX, NC, 0.125f);
    gemm_xwT<false,false,true,float><<<dim3(CDIM / 128, Mx / 128), thr256, 0, stream>>>(
        ATT, CDIM, ca_w, X1, CDIM, nullptr, X1, CDIM, CDIM);

    // --- MLP ---
    layernorm_bf16<<<Mx, thr256, 0, stream>>>(X1, ln2w, ln2b, XN);
    gemm_xwT<true,true,false,bf16><<<dim3(HID / 128, Mx / 128), thr256, 0, stream>>>(
        XN, CDIM, fc1_w, HB, HID, fc1_b, nullptr, 0, CDIM);
    gemm_xwT<true,false,true,float><<<dim3(CDIM / 128, Mx / 128), thr256, 0, stream>>>(
        HB, HID, fc2_w, out, CDIM, fc2_b, X1, CDIM, HID);
}
